// RankStatLoss_78271484002699
// MI455X (gfx1250) — matrix operand order repaired and checked by reading, not compiled, not run on hardware
//
#include <hip/hip_runtime.h>
#include <hip/hip_bf16.h>
#include <math.h>

#define N_ROWS 4096
#define D_FEAT 512
#define C_DIM  512
#define TOPK   5
#define BK     32

typedef __attribute__((ext_vector_type(16))) __bf16 v16bf;
typedef __attribute__((ext_vector_type(8)))  __bf16 v8bf;
typedef __attribute__((ext_vector_type(8)))  float  v8f;

typedef __attribute__((address_space(3))) __bf16  lds_bf16_t;
typedef __attribute__((address_space(3))) const v16bf lds_v16bf_t;

// ---------------- workspace layout (bytes) ----------------
#define WS_P1HI 0
#define WS_P1LO (WS_P1HI + (size_t)N_ROWS * C_DIM * 2)
#define WS_P2HI (WS_P1LO + (size_t)N_ROWS * C_DIM * 2)
#define WS_P2LO (WS_P2HI + (size_t)N_ROWS * C_DIM * 2)
#define WS_KEYS (WS_P2LO + (size_t)N_ROWS * C_DIM * 2)
#define WS_PART (WS_KEYS + (size_t)N_ROWS * 8)

// ---------------------------------------------------------------------------
// Kernel 1: split f32 probs into bf16 hi + bf16 lo (hi+lo ~ 16 mantissa bits)
// ---------------------------------------------------------------------------
__global__ void split_bf16_kernel(const float* __restrict__ p1,
                                  const float* __restrict__ p2,
                                  __bf16* __restrict__ p1hi, __bf16* __restrict__ p1lo,
                                  __bf16* __restrict__ p2hi, __bf16* __restrict__ p2lo) {
    size_t i = (size_t)blockIdx.x * blockDim.x + threadIdx.x;
    if (i >= (size_t)N_ROWS * C_DIM) return;
    float a = p1[i];
    __bf16 ah = (__bf16)a;
    p1hi[i] = ah;
    p1lo[i] = (__bf16)(a - (float)ah);
    float b = p2[i];
    __bf16 bh = (__bf16)b;
    p2hi[i] = bh;
    p2lo[i] = (__bf16)(b - (float)bh);
}

// ---------------------------------------------------------------------------
// Kernel 2: per-row top-5 indices of feat1, sorted ascending, packed in u64.
// ---------------------------------------------------------------------------
__global__ void topk_key_kernel(const float* __restrict__ feat,
                                unsigned long long* __restrict__ keys) {
    const int lane = threadIdx.x & 31;
    const int wave = threadIdx.x >> 5;
    const int row  = blockIdx.x * 8 + wave;
    const float* f = feat + (size_t)row * D_FEAT;

    float vals[16];
#pragma unroll
    for (int t = 0; t < 16; ++t) vals[t] = f[t * 32 + lane];

    int sel[TOPK];
#pragma unroll
    for (int p = 0; p < TOPK; ++p) {
        float bv = -INFINITY;
        int   bi = 0x7fffffff;
#pragma unroll
        for (int t = 0; t < 16; ++t) {
            int idx = t * 32 + lane;
            bool taken = false;
#pragma unroll
            for (int q = 0; q < TOPK; ++q)
                if (q < p) taken |= (sel[q] == idx);
            if (!taken && vals[t] > bv) { bv = vals[t]; bi = idx; }
        }
#pragma unroll
        for (int off = 16; off > 0; off >>= 1) {
            float ov = __shfl_xor(bv, off, 32);
            int   oi = __shfl_xor(bi, off, 32);
            if (ov > bv || (ov == bv && oi < bi)) { bv = ov; bi = oi; }
        }
        sel[p] = bi;
    }
#pragma unroll
    for (int a = 1; a < TOPK; ++a) {
        int v = sel[a]; int b = a - 1;
        while (b >= 0 && sel[b] > v) { sel[b + 1] = sel[b]; --b; }
        sel[b + 1] = v;
    }
    unsigned long long key = 0;
#pragma unroll
    for (int q = 0; q < TOPK; ++q)
        key |= ((unsigned long long)sel[q]) << (9 * q);
    if (lane == 0) keys[row] = key;
}

// ---------------------------------------------------------------------------
// Kernel 3: fused P = prob2 @ prob1^T (bf16 hi/lo WMMA) + BCE reduction.
// Block 256 thr = 8 waves; block tile 128x128; wave tile 32x64.
// B hi/lo tiles staged via double-buffered GLOBAL_LOAD_ASYNC_TO_LDS_B128
// (DMA of step k+1 overlaps WMMAs of step k); A loads stay direct global.
// LDS B layout per buffer: [col 0..127][hi 32 | lo 32] bf16 (16 KB/buffer).
// ---------------------------------------------------------------------------
__device__ inline v16bf load_fragA(const __bf16* __restrict__ base,
                                   int row, int k0, int laneHi) {
    const __bf16* p = base + (size_t)row * C_DIM + k0 + (laneHi << 4);
    v8bf c0 = *(const v8bf*)(p);
    v8bf c1 = *(const v8bf*)(p + 8);
    return __builtin_shufflevector(c0, c1, 0,1,2,3,4,5,6,7,8,9,10,11,12,13,14,15);
}

#define WMMA_BF16(A, B, Cacc) \
    __builtin_amdgcn_wmma_f32_16x16x32_bf16(false, (A), false, (B), (short)0, (Cacc), false, false)

__global__ void __launch_bounds__(256)
gemm_bce_kernel(const __bf16* __restrict__ p1hi, const __bf16* __restrict__ p1lo,
                const __bf16* __restrict__ p2hi, const __bf16* __restrict__ p2lo,
                const unsigned long long* __restrict__ keys,
                float* __restrict__ partials) {
    const int tid    = threadIdx.x;
    const int lane   = tid & 31;
    const int waveId = tid >> 5;
    const int wm     = waveId & 3;   // 0..3 : 32-row slices (M)
    const int wn     = waveId >> 2;  // 0..1 : 64-col slices (N)
    const int laneLo = lane & 15;
    const int laneHi = (lane >> 4) & 1;

    const int rowBase = blockIdx.y * 128 + wm * 32;  // j (prob2 rows)
    const int colBase = blockIdx.x * 128 + wn * 64;  // i (prob1 rows)
    const int blkCol  = blockIdx.x * 128;

    __shared__ __bf16 sB[2][128 * 64];   // 2 x 16 KB double buffer
    lds_bf16_t* sb3 = (lds_bf16_t*)&sB[0][0];

    // Stage one 128x32 hi+lo B slice (16 KB) into LDS buffer `buf`:
    // 1024 x 16B chunks, 4 async b128 loads per thread.
#define STAGE_B(buf, k0)                                                          \
    {                                                                             \
        _Pragma("unroll")                                                         \
        for (int q = 0; q < 4; ++q) {                                             \
            int c    = tid + 256 * q;          /* 0..1023 */                      \
            int col  = c >> 3;                                                    \
            int cc   = c & 7;                                                     \
            int hilo = cc >> 2;                                                   \
            int kk   = (cc & 3) << 3;          /* element offset 0,8,16,24 */     \
            const __bf16* src = (hilo ? p1lo : p1hi) +                            \
                (size_t)(blkCol + col) * C_DIM + (k0) + kk;                       \
            unsigned dst = (unsigned)(unsigned long long)                         \
                (sb3 + (buf) * (128 * 64) + col * 64 + hilo * 32 + kk);           \
            asm volatile("global_load_async_to_lds_b128 %0, %1, off"              \
                         :: "v"(dst), "v"(src) : "memory");                       \
        }                                                                         \
    }

    v8f acc[2][4] = {};

    STAGE_B(0, 0);

    for (int k0 = 0; k0 < C_DIM; k0 += BK) {
        const int buf = (k0 >> 5) & 1;
        asm volatile("s_wait_asynccnt 0x0" ::: "memory");
        __syncthreads();
        if (k0 + BK < C_DIM) STAGE_B(buf ^ 1, k0 + BK);

        // A fragments: direct global loads (16-bit A 16x32 layout)
        v16bf aHi[2], aLo[2];
#pragma unroll
        for (int mt = 0; mt < 2; ++mt) {
            const int r = rowBase + mt * 16 + laneLo;
            aHi[mt] = load_fragA(p2hi, r, k0, laneHi);
            aLo[mt] = load_fragA(p2lo, r, k0, laneHi);
        }
        if (k0 + BK < C_DIM)
            __builtin_prefetch(p2hi + (size_t)(rowBase + laneLo) * C_DIM + k0 + BK, 0, 3);

        // B fragments from LDS: lane n = column n, laneHi selects K 0-15/16-31
        v16bf bHi[4], bLo[4];
#pragma unroll
        for (int nt = 0; nt < 4; ++nt) {
            const int col = wn * 64 + nt * 16 + laneLo;   // column within block tile
            lds_bf16_t* pb = sb3 + buf * (128 * 64) + col * 64 + laneHi * 16;
            bHi[nt] = *(lds_v16bf_t*)(pb);
            bLo[nt] = *(lds_v16bf_t*)(pb + 32);
        }

#pragma unroll
        for (int mt = 0; mt < 2; ++mt) {
#pragma unroll
            for (int nt = 0; nt < 4; ++nt) {
                acc[mt][nt] = WMMA_BF16(aHi[mt], bHi[nt], acc[mt][nt]);
                acc[mt][nt] = WMMA_BF16(aHi[mt], bLo[nt], acc[mt][nt]);
                acc[mt][nt] = WMMA_BF16(aLo[mt], bHi[nt], acc[mt][nt]);
            }
        }
    }

    // ---- fused BCE epilogue: one hardware log per element ----
    unsigned long long kj[2][8];
#pragma unroll
    for (int mt = 0; mt < 2; ++mt)
#pragma unroll
        for (int e = 0; e < 8; ++e)
            kj[mt][e] = keys[rowBase + mt * 16 + laneHi * 8 + e];
    unsigned long long ki[4];
#pragma unroll
    for (int nt = 0; nt < 4; ++nt)
        ki[nt] = keys[colBase + nt * 16 + laneLo];

    float lsum = 0.0f;
#pragma unroll
    for (int mt = 0; mt < 2; ++mt) {
#pragma unroll
        for (int nt = 0; nt < 4; ++nt) {
#pragma unroll
            for (int e = 0; e < 8; ++e) {
                float p = acc[mt][nt][e];
                // t*log(p) + (1-t)*log(1-p)  ==  log(t ? p : 1-p)
                float x = (kj[mt][e] == ki[nt]) ? p : 1.0f - p;
                lsum += fmaxf(__logf(x), -100.0f);
            }
        }
    }

    __shared__ float sdata[256];
    sdata[tid] = lsum;
    __syncthreads();
#pragma unroll
    for (int s = 128; s > 0; s >>= 1) {
        if (tid < s) sdata[tid] += sdata[tid + s];
        __syncthreads();
    }
    if (tid == 0)
        partials[blockIdx.y * gridDim.x + blockIdx.x] = sdata[0];
}

// ---------------------------------------------------------------------------
// Kernel 4: deterministic final reduction of 1024 partials -> loss scalar.
// ---------------------------------------------------------------------------
__global__ void final_reduce_kernel(const float* __restrict__ partials,
                                    float* __restrict__ out) {
    __shared__ float sdata[256];
    const int tid = threadIdx.x;
    float s = partials[tid] + partials[tid + 256] + partials[tid + 512] + partials[tid + 768];
    sdata[tid] = s;
    __syncthreads();
#pragma unroll
    for (int k = 128; k > 0; k >>= 1) {
        if (tid < k) sdata[tid] += sdata[tid + k];
        __syncthreads();
    }
    if (tid == 0)
        out[0] = -sdata[0] * (1.0f / ((float)N_ROWS * (float)N_ROWS));
}

// ---------------------------------------------------------------------------
extern "C" void kernel_launch(void* const* d_in, const int* in_sizes, int n_in,
                              void* d_out, int out_size, void* d_ws, size_t ws_size,
                              hipStream_t stream) {
    const float* feat1 = (const float*)d_in[0];
    // d_in[1] = feat2 (unused by the loss)
    const float* prob1 = (const float*)d_in[2];
    const float* prob2 = (const float*)d_in[3];

    char* ws = (char*)d_ws;
    __bf16* p1hi = (__bf16*)(ws + WS_P1HI);
    __bf16* p1lo = (__bf16*)(ws + WS_P1LO);
    __bf16* p2hi = (__bf16*)(ws + WS_P2HI);
    __bf16* p2lo = (__bf16*)(ws + WS_P2LO);
    unsigned long long* keys = (unsigned long long*)(ws + WS_KEYS);
    float* partials = (float*)(ws + WS_PART);
    float* out = (float*)d_out;

    {
        const size_t total = (size_t)N_ROWS * C_DIM;
        dim3 grid((unsigned)((total + 255) / 256)), block(256);
        split_bf16_kernel<<<grid, block, 0, stream>>>(prob1, prob2, p1hi, p1lo, p2hi, p2lo);
    }
    {
        dim3 grid(N_ROWS / 8), block(256);
        topk_key_kernel<<<grid, block, 0, stream>>>(feat1, keys);
    }
    {
        dim3 grid(N_ROWS / 128, N_ROWS / 128), block(256);
        gemm_bce_kernel<<<grid, block, 0, stream>>>(p1hi, p1lo, p2hi, p2lo, keys, partials);
    }
    {
        final_reduce_kernel<<<1, 256, 0, stream>>>(partials, out);
    }
}
